// DeepKcat_13615046328905
// MI455X (gfx1250) — matrix operand order repaired and checked by reading, not compiled, not run on hardware
//
#include <hip/hip_runtime.h>
#include <hip/hip_bf16.h>

// ---------------------------------------------------------------------------
// Types / WMMA helpers (gfx1250, wave32)
// ---------------------------------------------------------------------------
typedef float  v8f   __attribute__((ext_vector_type(8)));
typedef __bf16 v16bf __attribute__((ext_vector_type(16)));
typedef __bf16 v8bf  __attribute__((ext_vector_type(8)));

__device__ __forceinline__ v8f vzero8() {
  v8f z = {0.f,0.f,0.f,0.f,0.f,0.f,0.f,0.f};
  return z;
}

// A-fragment (16x32 bf16, M x K): lane = M row; lanes 0-15 hold K {0..7,16..23},
// lanes 16-31 hold K {8..15,24..31}.  Two contiguous 16B loads from row-major A.
__device__ __forceinline__ v16bf load_fragA(const __bf16* __restrict__ row, int k0, int half) {
  v8bf a = *(const v8bf*)(row + k0 + 8*half);
  v8bf b = *(const v8bf*)(row + k0 + 16 + 8*half);
  v16bf r;
#pragma unroll
  for (int i = 0; i < 8; ++i) { r[i] = a[i]; r[i+8] = b[i]; }
  return r;
}

// B-fragment (32x16 bf16, K x N): lane = N column; lanes 0-15 hold K {0..7,16..23},
// lanes 16-31 hold K {8..15,24..31}.  Two contiguous 16B loads from row-major B^T.
__device__ __forceinline__ v16bf load_fragB(const __bf16* __restrict__ row, int k0, int half) {
  v8bf a = *(const v8bf*)(row + k0 + 8*half);
  v8bf b = *(const v8bf*)(row + k0 + 16 + 8*half);
  v16bf r;
#pragma unroll
  for (int i = 0; i < 8; ++i) { r[i] = a[i]; r[i+8] = b[i]; }
  return r;
}

__device__ __forceinline__ v8f wmma_bf16(v16bf a, v16bf b, v8f c) {
  return __builtin_amdgcn_wmma_f32_16x16x32_bf16(false, a, false, b, (short)0, c, false, false);
}

// ---------------------------------------------------------------------------
// Generic WMMA GEMM:  C[M, N] = A[M,K](bf16) @ Bt[N,K](bf16)^T  (+bias, relu)
// One wave computes a 16 x (16*NT) tile: the A fragment is loaded once per
// K-chunk and reused across NT B fragments (register blocking along N).
// M multiple of 16, N multiple of 16*NT, K multiple of 32.
// ---------------------------------------------------------------------------
template <int NT>
__global__ __launch_bounds__(32, 1)
void gemm16t(const __bf16* __restrict__ A, int lda,
             const __bf16* __restrict__ Bt, int ldb,
             float* __restrict__ Cf, __bf16* __restrict__ Cb, int ldc,
             const float* __restrict__ bias, int nbias, int relu, int K) {
  const int lane = threadIdx.x & 31;
  const int half = lane >> 4, l16 = lane & 15;
  const __bf16* pa = A + (long)(blockIdx.y * 16 + l16) * lda;
  const __bf16* pb[NT];
#pragma unroll
  for (int nt = 0; nt < NT; ++nt)
    pb[nt] = Bt + (long)(blockIdx.x * (16 * NT) + nt * 16 + l16) * ldb;
  v8f acc[NT];
#pragma unroll
  for (int nt = 0; nt < NT; ++nt) acc[nt] = vzero8();

  for (int k0 = 0; k0 < K; k0 += 32) {
    if (k0 + 64 <= K) {  // speculative prefetch of next K tile -> global_prefetch_b8
      __builtin_prefetch(pa + k0 + 32, 0, 1);
#pragma unroll
      for (int nt = 0; nt < NT; ++nt) __builtin_prefetch(pb[nt] + k0 + 32, 0, 1);
    }
    v16bf af = load_fragA(pa, k0, half);
#pragma unroll
    for (int nt = 0; nt < NT; ++nt) {
      v16bf bf = load_fragB(pb[nt], k0, half);
      acc[nt] = wmma_bf16(af, bf, acc[nt]);
    }
  }
#pragma unroll
  for (int nt = 0; nt < NT; ++nt) {
    const int col = blockIdx.x * (16 * NT) + nt * 16 + l16;
#pragma unroll
    for (int r = 0; r < 8; ++r) {
      long row = blockIdx.y * 16 + r + 8 * half;   // D layout: elem r -> M = r + 8*half
      float v = acc[nt][r];
      if (bias && col < nbias) v += bias[col];
      if (relu) v = fmaxf(v, 0.f);
      long idx = row * (long)ldc + col;
      if (Cf) Cf[idx] = v;
      if (Cb) Cb[idx] = (__bf16)v;
    }
  }
}

// ---------------------------------------------------------------------------
// Flash attention (one wave per (head, 16-row q tile)), dh=128, 4 heads, D=512.
// Q,K: bf16 [L,512] row-major.  Vt: bf16 [512,L] (row = head*128+d).  O: f32 [L,512].
// __launch_bounds__(32,1): single-wave workgroup, full register budget -> no spills.
// ---------------------------------------------------------------------------
__global__ __launch_bounds__(32, 1)
void flash_attn(const __bf16* __restrict__ Q, const __bf16* __restrict__ Km,
                const __bf16* __restrict__ Vt, float* __restrict__ O, int L) {
  const int DD = 512, DH = 128;
  const int lane = threadIdx.x & 31, half = lane >> 4, l16 = lane & 15;
  const int head = blockIdx.y;
  const int q0 = blockIdx.x * 16;
  __shared__ float sP[16 * 32];

  const __bf16* qrow = Q + (long)(q0 + l16) * DD + head * DH;
  v16bf qf[4];
#pragma unroll
  for (int c = 0; c < 4; ++c) qf[c] = load_fragA(qrow, c * 32, half);

  v8f o[8];
#pragma unroll
  for (int t = 0; t < 8; ++t) o[t] = vzero8();
  float m[8], lsum[8];
#pragma unroll
  for (int r = 0; r < 8; ++r) { m[r] = -3.0e38f; lsum[r] = 0.f; }

  const float scale = 0.0883883476483184f;  // 1/sqrt(128)

  for (int kb = 0; kb < L; kb += 32) {
    // S = Q @ K^T  (two 16-key column tiles); B-frag of K^T == row-major rows of K
    v8f s0 = vzero8(), s1 = vzero8();
    const __bf16* k0p = Km + (long)(kb + l16) * DD + head * DH;
    const __bf16* k1p = Km + (long)(kb + 16 + l16) * DD + head * DH;
#pragma unroll
    for (int c = 0; c < 4; ++c) {
      v16bf kf0 = load_fragB(k0p, c * 32, half);
      v16bf kf1 = load_fragB(k1p, c * 32, half);
      s0 = wmma_bf16(qf[c], kf0, s0);
      s1 = wmma_bf16(qf[c], kf1, s1);
    }
    // online softmax per row (row r+8*half lives in the 16 lanes of this half-wave)
    float fs[8];
#pragma unroll
    for (int r = 0; r < 8; ++r) {
      float a = s0[r] * scale, b = s1[r] * scale;
      float mx = fmaxf(a, b);
      mx = fmaxf(mx, __shfl_xor(mx, 1, 32));
      mx = fmaxf(mx, __shfl_xor(mx, 2, 32));
      mx = fmaxf(mx, __shfl_xor(mx, 4, 32));
      mx = fmaxf(mx, __shfl_xor(mx, 8, 32));
      float nm = fmaxf(m[r], mx);
      fs[r] = __expf(m[r] - nm);
      m[r] = nm;
      float p0 = __expf(a - nm), p1 = __expf(b - nm);
      s0[r] = p0; s1[r] = p1;
      float rs = p0 + p1;
      rs += __shfl_xor(rs, 1, 32);
      rs += __shfl_xor(rs, 2, 32);
      rs += __shfl_xor(rs, 4, 32);
      rs += __shfl_xor(rs, 8, 32);
      lsum[r] = lsum[r] * fs[r] + rs;
    }
#pragma unroll
    for (int t = 0; t < 8; ++t)
#pragma unroll
      for (int r = 0; r < 8; ++r) o[t][r] *= fs[r];

    // transpose P (D-layout -> A-layout) through LDS
    __syncthreads();
#pragma unroll
    for (int r = 0; r < 8; ++r) {
      sP[(r + 8 * half) * 32 + l16] = s0[r];
      sP[(r + 8 * half) * 32 + 16 + l16] = s1[r];
    }
    __syncthreads();
    v16bf pf;
#pragma unroll
    for (int i = 0; i < 8; ++i) {
      pf[i]     = (__bf16)sP[l16 * 32 + 8 * half + i];
      pf[i + 8] = (__bf16)sP[l16 * 32 + 16 + 8 * half + i];
    }
    // O += P @ V  (8 output-column tiles of 16)
#pragma unroll
    for (int t = 0; t < 8; ++t) {
      const __bf16* vrow = Vt + (long)(head * DH + t * 16 + l16) * L + kb;
      v16bf vf = load_fragB(vrow, 0, half);
      o[t] = wmma_bf16(pf, vf, o[t]);
    }
  }
#pragma unroll
  for (int t = 0; t < 8; ++t)
#pragma unroll
    for (int r = 0; r < 8; ++r) {
      long row = q0 + r + 8 * half;
      int col = head * DH + t * 16 + l16;
      O[row * (long)DD + col] = o[t][r] / lsum[r];
    }
}

// ---------------------------------------------------------------------------
// Elementwise / utility kernels
// ---------------------------------------------------------------------------
// dst[i*dC+j] = (i<sR && j<sC) ? src[i*lds+j] : 0    (f32 -> bf16, padded)
__global__ void conv_bf16(__bf16* __restrict__ dst, const float* __restrict__ src,
                          int dR, int dC, int sR, int sC, int lds) {
  long gid = (long)blockIdx.x * blockDim.x + threadIdx.x;
  long n = (long)dR * dC;
  if (gid >= n) return;
  int i = (int)(gid / dC), j = (int)(gid % dC);
  float v = (i < sR && j < sC) ? src[(long)i * lds + j] : 0.f;
  dst[gid] = (__bf16)v;
}

// dst[i*dC+j] = (i<sC && j<sR) ? src[j*lds+i] : 0    (transpose f32 -> bf16, padded)
__global__ void convT_bf16(__bf16* __restrict__ dst, const float* __restrict__ src,
                           int dR, int dC, int sR, int sC, int lds) {
  long gid = (long)blockIdx.x * blockDim.x + threadIdx.x;
  long n = (long)dR * dC;
  if (gid >= n) return;
  int i = (int)(gid / dC), j = (int)(gid % dC);
  float v = (i < sC && j < sR) ? src[(long)j * lds + i] : 0.f;
  dst[gid] = (__bf16)v;
}

__global__ void scatter_adj(const int* __restrict__ ei, int ne, int n, __bf16* __restrict__ adj) {
  int e = blockIdx.x * blockDim.x + threadIdx.x;
  if (e >= ne) return;
  int a = ei[e], b = ei[ne + e];
  adj[(long)a * n + b] = (__bf16)1.0f;
}

// Diffusion categorical sample -> one-hot transposed bf16 [32pad, n] (buffer pre-zeroed)
__global__ void diffuse_onehotT(const float* __restrict__ x, const int* __restrict__ tptr,
                                __bf16* __restrict__ XtT, int n) {
  int i = blockIdx.x * blockDim.x + threadIdx.x;
  if (i >= n) return;
  const float s = 0.008f;
  float tf = (float)tptr[0] / 500.0f;
  const float PIH = 1.5707963267948966f;
  float c1 = __cosf((tf + s) / (1.f + s) * PIH);
  float c0 = __cosf(s / (1.f + s) * PIH);
  float ab = (c1 * c1) / (c0 * c0);
  float best = -3e38f; int cls = 0;
  for (int k = 0; k < 20; ++k) {
    float p = ab * x[i * 20 + k] + (1.f - ab) * 0.05f;
    unsigned z = (unsigned)(i * 20 + k + 1);
    z *= 0x9E3779B9u; z ^= z >> 16; z *= 0x85EBCA6Bu; z ^= z >> 13; z *= 0xC2B2AE35u; z ^= z >> 16;
    float u = ((float)(z >> 8) + 0.5f) * (1.0f / 16777216.0f);
    float g = -__logf(-__logf(u));
    float sc = __logf(p + 1e-9f) + g;
    if (sc > best) { best = sc; cls = k; }
  }
  XtT[(long)cls * n + i] = (__bf16)1.0f;
}

__global__ void gather_fpT(const float* __restrict__ emb, const int* __restrict__ idx,
                           __bf16* __restrict__ dst, int n) {
  int gid = blockIdx.x * blockDim.x + threadIdx.x;
  if (gid >= n * 20) return;
  int node = gid / 20, d = gid % 20;
  dst[(long)d * n + node] = (__bf16)emb[(long)idx[node] * 20 + d];
}

__global__ void gather_pe(const float* __restrict__ emb, const int* __restrict__ words,
                          float* __restrict__ h, __bf16* __restrict__ hb, int L) {
  long gid = (long)blockIdx.x * blockDim.x + threadIdx.x;
  if (gid >= (long)L * 512) return;
  int nIdx = (int)(gid / 512), d = (int)(gid % 512);
  float v = emb[(long)words[nIdx] * 512 + d];
  int i2 = d >> 1;
  float ang = (float)nIdx * __powf(10000.0f, -(float)(2 * i2) * (1.0f / 512.0f));
  v += (d & 1) ? __cosf(ang) : __sinf(ang);
  h[gid] = v; hb[gid] = (__bf16)v;
}

// y = LN(x + r) * g + b ; writes f32 (in-place on x allowed) and bf16.  D=512, block=256.
__global__ void ln_res(const float* __restrict__ x, const float* __restrict__ r,
                       const float* __restrict__ g, const float* __restrict__ b,
                       float* __restrict__ yf, __bf16* __restrict__ yb) {
  const int row = blockIdx.x, t = threadIdx.x;
  __shared__ float red[256];
  long base = (long)row * 512;
  float v0 = x[base + t] + r[base + t];
  float v1 = x[base + 256 + t] + r[base + 256 + t];
  red[t] = v0 + v1; __syncthreads();
  for (int st = 128; st > 0; st >>= 1) { if (t < st) red[t] += red[t + st]; __syncthreads(); }
  float mean = red[0] * (1.0f / 512.0f); __syncthreads();
  float d0 = v0 - mean, d1 = v1 - mean;
  red[t] = d0 * d0 + d1 * d1; __syncthreads();
  for (int st = 128; st > 0; st >>= 1) { if (t < st) red[t] += red[t + st]; __syncthreads(); }
  float inv = rsqrtf(red[0] * (1.0f / 512.0f) + 1e-5f);
  float y0 = d0 * inv * g[t] + b[t];
  float y1 = d1 * inv * g[256 + t] + b[256 + t];
  yf[base + t] = y0; yf[base + 256 + t] = y1;
  yb[base + t] = (__bf16)y0; yb[base + 256 + t] = (__bf16)y1;
}

__global__ void col_mean(const float* __restrict__ src, int R, int ld, float* __restrict__ out) {
  const int c = blockIdx.x, t = threadIdx.x;
  __shared__ float red[256];
  float s = 0.f;
  for (int r = t; r < R; r += 256) s += src[(long)r * ld + c];
  red[t] = s; __syncthreads();
  for (int st = 128; st > 0; st >>= 1) { if (t < st) red[t] += red[t + st]; __syncthreads(); }
  if (t == 0) out[c] = red[0] / (float)R;
}

// fusion head: cat=[sub,prot,seq,time] (80), 3x (relu(cat)@Wout[j]+bout[j]), final relu.
__global__ void fusion(const float* __restrict__ sub, const float* __restrict__ prot,
                       const float* __restrict__ seq, const float* __restrict__ Wtime,
                       const float* __restrict__ btime, const float* __restrict__ Wout,
                       const float* __restrict__ bout, const int* __restrict__ tptr,
                       float* __restrict__ out) {
  __shared__ float cat[80], nxt[80];
  int tid = threadIdx.x;
  float tf = (float)tptr[0] / 500.0f;
  if (tid < 20) {
    cat[tid]      = sub[tid];
    cat[20 + tid] = prot[tid];
    cat[40 + tid] = seq[tid];
    cat[60 + tid] = tf * Wtime[tid] + btime[tid];
  }
  __syncthreads();
  for (int j = 0; j < 3; ++j) {
    if (tid < 80) {
      float acc = bout[j * 80 + tid];
      for (int k = 0; k < 80; ++k) acc += fmaxf(cat[k], 0.f) * Wout[j * 6400 + k * 80 + tid];
      nxt[tid] = acc;
    }
    __syncthreads();
    if (tid < 80) cat[tid] = nxt[tid];
    __syncthreads();
  }
  if (tid < 80) out[tid] = fmaxf(cat[tid], 0.f);
  if (tid < 20) out[80 + tid] = seq[tid];
}

// ---------------------------------------------------------------------------
// Host orchestration
// ---------------------------------------------------------------------------
extern "C" void kernel_launch(void* const* d_in, const int* in_sizes, int n_in,
                              void* d_out, int out_size, void* d_ws, size_t ws_size,
                              hipStream_t stream) {
  (void)n_in; (void)out_size; (void)ws_size;
  const float* x       = (const float*)d_in[0];
  const float* smile   = (const float*)d_in[1];
  const float* emb_fp  = (const float*)d_in[2];
  const float* emb_word= (const float*)d_in[3];
  const float* gW1 = (const float*)d_in[4];  const float* gb1 = (const float*)d_in[5];
  const float* gW2 = (const float*)d_in[6];  const float* gb2 = (const float*)d_in[7];
  const float* gW3 = (const float*)d_in[8];  const float* gb3 = (const float*)d_in[9];
  const float* Wq  = (const float*)d_in[10]; const float* bq  = (const float*)d_in[11];
  const float* Wk  = (const float*)d_in[12]; const float* bk  = (const float*)d_in[13];
  const float* Wv  = (const float*)d_in[14]; const float* bv  = (const float*)d_in[15];
  const float* Wo  = (const float*)d_in[16]; const float* bo  = (const float*)d_in[17];
  const float* ln1g= (const float*)d_in[18]; const float* ln1b= (const float*)d_in[19];
  const float* Wff1= (const float*)d_in[20]; const float* bff1= (const float*)d_in[21];
  const float* Wff2= (const float*)d_in[22]; const float* bff2= (const float*)d_in[23];
  const float* ln2g= (const float*)d_in[24]; const float* ln2b= (const float*)d_in[25];
  const float* Wproj=(const float*)d_in[26]; const float* bproj=(const float*)d_in[27];
  const float* Wout= (const float*)d_in[28]; const float* bout= (const float*)d_in[29];
  const float* Wtime=(const float*)d_in[30]; const float* btime=(const float*)d_in[31];
  const int* fps   = (const int*)d_in[32];
  const int* words = (const int*)d_in[33];
  const int* edges = (const int*)d_in[35];
  const int* tptr  = (const int*)d_in[36];

  const int L  = in_sizes[33];          // 4096 tokens
  const int NR = in_sizes[0] / 20;      // 4096 residues
  const int NA = in_sizes[32];          // 256 atoms
  const int NE = in_sizes[35] / 2;      // 65536 edges
  const int D = 512, FF = 2048, H1 = 128, H2 = 64;

  size_t off = 0;
  auto alloc = [&](size_t bytes) -> void* {
    void* p = (char*)d_ws + off;
    off += (bytes + 255) & ~(size_t)255;
    return p;
  };
  // bf16 scratch
  __bf16* hbf    = (__bf16*)alloc((size_t)L * D * 2);
  __bf16* qbf    = (__bf16*)alloc((size_t)L * D * 2);
  __bf16* kbf    = (__bf16*)alloc((size_t)L * D * 2);
  __bf16* vtbf   = (__bf16*)alloc((size_t)D * L * 2);
  __bf16* attnbf = (__bf16*)alloc((size_t)L * D * 2);
  __bf16* ff1bf  = (__bf16*)alloc((size_t)L * FF * 2);
  __bf16* wqt    = (__bf16*)alloc((size_t)2 * D * D * 2);
  __bf16* wkt    = (__bf16*)alloc((size_t)2 * D * D * 2);
  __bf16* wvt    = (__bf16*)alloc((size_t)2 * D * D * 2);
  __bf16* wot    = (__bf16*)alloc((size_t)2 * D * D * 2);
  __bf16* wff1t  = (__bf16*)alloc((size_t)2 * FF * D * 2);
  __bf16* wff2t  = (__bf16*)alloc((size_t)2 * D * FF * 2);
  __bf16* wprojt = (__bf16*)alloc((size_t)32 * D * 2);
  __bf16* gw1t   = (__bf16*)alloc((size_t)H1 * 32 * 2);
  __bf16* gw2t   = (__bf16*)alloc((size_t)H2 * H1 * 2);
  __bf16* gw3t   = (__bf16*)alloc((size_t)32 * H2 * 2);
  __bf16* adjP   = (__bf16*)alloc((size_t)NR * NR * 2);
  __bf16* adjS   = (__bf16*)alloc((size_t)NA * NA * 2);
  __bf16* xtP    = (__bf16*)alloc((size_t)32 * NR * 2);
  __bf16* xtS    = (__bf16*)alloc((size_t)32 * NA * 2);
  __bf16* axbf   = (__bf16*)alloc((size_t)NR * 32 * 2);
  __bf16* h1tb   = (__bf16*)alloc((size_t)H1 * NR * 2);
  __bf16* ah1b   = (__bf16*)alloc((size_t)NR * H1 * 2);
  __bf16* h2tb   = (__bf16*)alloc((size_t)H2 * NR * 2);
  __bf16* ah2b   = (__bf16*)alloc((size_t)NR * H2 * 2);
  // f32 scratch
  float* hf    = (float*)alloc((size_t)L * D * 4);
  float* vf    = (float*)alloc((size_t)L * D * 4);
  float* attnf = (float*)alloc((size_t)L * D * 4);
  float* t0f   = (float*)alloc((size_t)L * D * 4);
  float* projf = (float*)alloc((size_t)L * 32 * 4);
  float* axf   = (float*)alloc((size_t)NR * 32 * 4);
  float* h1f   = (float*)alloc((size_t)NR * H1 * 4);
  float* ah1f  = (float*)alloc((size_t)NR * H1 * 4);
  float* h2f   = (float*)alloc((size_t)NR * H2 * 4);
  float* ah2f  = (float*)alloc((size_t)NR * H2 * 4);
  float* goutf = (float*)alloc((size_t)NR * 32 * 4);
  float* sub20 = (float*)alloc(32 * 4);
  float* prot20= (float*)alloc(32 * 4);
  float* seq20 = (float*)alloc(32 * 4);

  auto gemm = [&](const __bf16* A, int lda, const __bf16* Bt, int ldb,
                  float* Cf, __bf16* Cb, int ldc, const float* bias, int nbias,
                  int relu, int M, int N, int K) {
    if ((N & 63) == 0) {
      dim3 grid(N / 64, M / 16);
      gemm16t<4><<<grid, 32, 0, stream>>>(A, lda, Bt, ldb, Cf, Cb, ldc, bias, nbias, relu, K);
    } else if ((N & 31) == 0) {
      dim3 grid(N / 32, M / 16);
      gemm16t<2><<<grid, 32, 0, stream>>>(A, lda, Bt, ldb, Cf, Cb, ldc, bias, nbias, relu, K);
    } else {
      dim3 grid(N / 16, M / 16);
      gemm16t<1><<<grid, 32, 0, stream>>>(A, lda, Bt, ldb, Cf, Cb, ldc, bias, nbias, relu, K);
    }
  };
  auto convN = [&](__bf16* dst, const float* src, int dR, int dC, int sR, int sC, int lds) {
    long n = (long)dR * dC;
    conv_bf16<<<(unsigned)((n + 255) / 256), 256, 0, stream>>>(dst, src, dR, dC, sR, sC, lds);
  };
  auto convT = [&](__bf16* dst, const float* src, int dR, int dC, int sR, int sC, int lds) {
    long n = (long)dR * dC;
    convT_bf16<<<(unsigned)((n + 255) / 256), 256, 0, stream>>>(dst, src, dR, dC, sR, sC, lds);
  };

  // ---- Stage 0: weight transposes (f32 -> bf16, B^T layout) ----
  for (int l = 0; l < 2; ++l) {
    convT(wqt + (size_t)l * D * D, Wq + (size_t)l * D * D, D, D, D, D, D);
    convT(wkt + (size_t)l * D * D, Wk + (size_t)l * D * D, D, D, D, D, D);
    convT(wvt + (size_t)l * D * D, Wv + (size_t)l * D * D, D, D, D, D, D);
    convT(wot + (size_t)l * D * D, Wo + (size_t)l * D * D, D, D, D, D, D);
    convT(wff1t + (size_t)l * FF * D, Wff1 + (size_t)l * D * FF, FF, D, D, FF, FF);
    convT(wff2t + (size_t)l * D * FF, Wff2 + (size_t)l * FF * D, D, FF, FF, D, D);
  }
  convT(wprojt, Wproj, 32, D, D, 20, 20);
  convT(gw1t, gW1, H1, 32, 20, H1, H1);
  convT(gw2t, gW2, H2, H1, H1, H2, H2);
  convT(gw3t, gW3, 32, H2, H2, 20, 20);

  // ---- Stage 1: diffusion one-hot X^T (bf16, zero-padded to 32 rows) ----
  convN(xtP, x, 32, NR, 0, 0, 0);  // zero fill
  diffuse_onehotT<<<(NR + 255) / 256, 256, 0, stream>>>(x, tptr, xtP, NR);

  // ---- Stage 2: protein adjacency (bf16 0/1 via idempotent stores) ----
  convN(adjP, x, NR, NR, 0, 0, 0); // zero fill
  scatter_adj<<<(NE + 255) / 256, 256, 0, stream>>>(edges, NE, NR, adjP);

  // ---- Stage 3: substrate inputs ----
  convN(adjS, smile, NA, NA, NA, NA, NA);
  convN(xtS, x, 32, NA, 0, 0, 0);  // zero fill
  gather_fpT<<<(NA * 20 + 255) / 256, 256, 0, stream>>>(emb_fp, fps, xtS, NA);

  // ---- shared 3-layer GCN ----
  auto run_gcn = [&](const __bf16* adj, int n, const __bf16* xt, float* outv) {
    gemm(adj, n, xt, n, axf, nullptr, 32, nullptr, 0, 0, n, 32, n);          // A@X
    convN(axbf, axf, n, 32, n, 32, 32);
    gemm(axbf, 32, gw1t, 32, h1f, nullptr, H1, gb1, H1, 1, n, H1, 32);       // relu(.@W1+b1)
    convT(h1tb, h1f, H1, n, n, H1, H1);
    gemm(adj, n, h1tb, n, ah1f, nullptr, H1, nullptr, 0, 0, n, H1, n);       // A@h
    convN(ah1b, ah1f, n, H1, n, H1, H1);
    gemm(ah1b, H1, gw2t, H1, h2f, nullptr, H2, gb2, H2, 1, n, H2, H1);       // relu(.@W2+b2)
    convT(h2tb, h2f, H2, n, n, H2, H2);
    gemm(adj, n, h2tb, n, ah2f, nullptr, H2, nullptr, 0, 0, n, H2, n);       // A@h
    convN(ah2b, ah2f, n, H2, n, H2, H2);
    gemm(ah2b, H2, gw3t, H2, goutf, nullptr, 32, gb3, 20, 0, n, 32, H2);     // .@W3+b3
    col_mean<<<20, 256, 0, stream>>>(goutf, n, 32, outv);
  };
  run_gcn(adjS, NA, xtS, sub20);   // substrate
  run_gcn(adjP, NR, xtP, prot20);  // protein

  // ---- Stage 4: transformer ----
  gather_pe<<<(unsigned)(((long)L * D + 255) / 256), 256, 0, stream>>>(emb_word, words, hf, hbf, L);
  for (int l = 0; l < 2; ++l) {
    const __bf16* wq_l = wqt + (size_t)l * D * D;
    const __bf16* wk_l = wkt + (size_t)l * D * D;
    const __bf16* wv_l = wvt + (size_t)l * D * D;
    const __bf16* wo_l = wot + (size_t)l * D * D;
    gemm(hbf, D, wq_l, D, nullptr, qbf, D, bq + l * D, D, 0, L, D, D);
    gemm(hbf, D, wk_l, D, nullptr, kbf, D, bk + l * D, D, 0, L, D, D);
    gemm(hbf, D, wv_l, D, vf, nullptr, D, bv + l * D, D, 0, L, D, D);
    convT(vtbf, vf, D, L, L, D, D);                       // V^T per head (bf16)
    flash_attn<<<dim3(L / 16, 4), 32, 0, stream>>>(qbf, kbf, vtbf, attnf, L);
    convN(attnbf, attnf, L, D, L, D, D);
    gemm(attnbf, D, wo_l, D, t0f, nullptr, D, bo + l * D, D, 0, L, D, D);
    ln_res<<<L, 256, 0, stream>>>(hf, t0f, ln1g + l * D, ln1b + l * D, hf, hbf);
    gemm(hbf, D, wff1t + (size_t)l * FF * D, D, nullptr, ff1bf, FF, bff1 + l * FF, FF, 1, L, FF, D);
    gemm(ff1bf, FF, wff2t + (size_t)l * D * FF, FF, t0f, nullptr, D, bff2 + l * D, D, 0, L, D, FF);
    ln_res<<<L, 256, 0, stream>>>(hf, t0f, ln2g + l * D, ln2b + l * D, hf, hbf);
  }
  gemm(hbf, D, wprojt, D, projf, nullptr, 32, bproj, 20, 0, L, 32, D);
  col_mean<<<20, 256, 0, stream>>>(projf, L, 32, seq20);

  // ---- Stage 5: fusion head + outputs ----
  fusion<<<1, 128, 0, stream>>>(sub20, prot20, seq20, Wtime, btime, Wout, bout, tptr,
                                (float*)d_out);
}
